// R3MRecurrentLM_19662360281132
// MI455X (gfx1250) — compile-verified, matrix operand order checked by reading, not run on hardware
//
#include <hip/hip_runtime.h>
#include <math.h>

typedef _Float16 f16;
typedef _Float16 v16h __attribute__((ext_vector_type(16)));
typedef _Float16 v8h  __attribute__((ext_vector_type(8)));
typedef float    v8f  __attribute__((ext_vector_type(8)));

static constexpr int   BB = 2048;
static constexpr int   DD = 512;
static constexpr int   SS = 128;
static constexpr int   EE = 4;
static constexpr int   HH = 2048;
static constexpr int   KSTEPS = 4;
static constexpr float ETA_ = 0.1f;
static constexpr float GATE_MAX_ = 0.2f;
static constexpr int   SINK_ITERS_ = 8;
static constexpr float EPS_ = 1e-6f;

// ---------------- wave helpers (wave32) ----------------
__device__ __forceinline__ float wsum32(float v) {
#pragma unroll
  for (int m = 16; m; m >>= 1) v += __shfl_xor(v, m, 32);
  return v;
}
__device__ __forceinline__ float wmax32(float v) {
#pragma unroll
  for (int m = 16; m; m >>= 1) v = fmaxf(v, __shfl_xor(v, m, 32));
  return v;
}
__device__ __forceinline__ float gelu_exact(float x) {
  return 0.5f * x * (1.f + erff(x * 0.70710678118654752f));
}

// load 16 halves for one WMMA fragment line: K-interleave {kb..kb+7, kb+16..kb+23}
__device__ __forceinline__ v16h ld_frag(const f16* __restrict__ p, int kb) {
  v8h lo = *(const v8h*)(p + kb);
  v8h hi = *(const v8h*)(p + kb + 16);
  v16h r;
#pragma unroll
  for (int i = 0; i < 8; ++i) { r[i] = lo[i]; r[i + 8] = hi[i]; }
  return r;
}

// ---------------- generic WMMA GEMM, 2x2 register-blocked (32x32 per wave) ----
// C(MxN) = A(MxK, f16 row-major) * Bt(NxK, f16 "transposed": column-major weights)
// mode 0: Cf = acc + bias
// mode 1: Cf (+)= rowScale[row*rss] * acc      (dense MoE accumulation)
// mode 2: Ch = f16(gelu(acc))                  (MLP hidden, feeds next GEMM as A)
// mode 3: Ch = f16(acc)
__global__ void gemm_wmma(const f16* __restrict__ A, const f16* __restrict__ Bt,
                          float* __restrict__ Cf, f16* __restrict__ Ch,
                          const float* __restrict__ biasN,
                          const float* __restrict__ rowScale, int rss,
                          int M, int N, int K, int mode, int accumulate) {
  int tilesN = N >> 5;                  // 32-wide super-tiles
  int totalTiles = (M >> 5) * tilesN;
  int gw = blockIdx.x * 8 + (threadIdx.x >> 5);
  if (gw >= totalTiles) return;
  int tM = gw / tilesN, tN = gw % tilesN;
  int lane = threadIdx.x & 31;
  int hi = lane >> 4;                   // lane half selects K sub-block
  int lr = lane & 15;
  const f16* Arow0 = A + (size_t)(tM * 32 + lr) * K;
  const f16* Arow1 = Arow0 + (size_t)16 * K;
  const f16* Bcol0 = Bt + (size_t)(tN * 32 + lr) * K;
  const f16* Bcol1 = Bcol0 + (size_t)16 * K;
  v8f acc00 = {}, acc01 = {}, acc10 = {}, acc11 = {};
  for (int k0 = 0; k0 < K; k0 += 32) {
    int kb = k0 + hi * 8;
    v16h a0 = ld_frag(Arow0, kb);
    v16h a1 = ld_frag(Arow1, kb);
    v16h b0 = ld_frag(Bcol0, kb);
    v16h b1 = ld_frag(Bcol1, kb);
    acc00 = __builtin_amdgcn_wmma_f32_16x16x32_f16(false, a0, false, b0, (short)0, acc00, false, false);
    acc01 = __builtin_amdgcn_wmma_f32_16x16x32_f16(false, a0, false, b1, (short)0, acc01, false, false);
    acc10 = __builtin_amdgcn_wmma_f32_16x16x32_f16(false, a1, false, b0, (short)0, acc10, false, false);
    acc11 = __builtin_amdgcn_wmma_f32_16x16x32_f16(false, a1, false, b1, (short)0, acc11, false, false);
  }
#pragma unroll
  for (int si = 0; si < 2; ++si) {
#pragma unroll
    for (int sj = 0; sj < 2; ++sj) {
      v8f acc = (si == 0) ? ((sj == 0) ? acc00 : acc01)
                          : ((sj == 0) ? acc10 : acc11);
      int col = tN * 32 + sj * 16 + lr;
      float bias = biasN ? biasN[col] : 0.f;
#pragma unroll
      for (int rr = 0; rr < 8; ++rr) {
        int row = tM * 32 + si * 16 + (hi ? 8 + rr : rr);
        float v = acc[rr] + bias;
        size_t idx = (size_t)row * N + col;
        if (mode == 0) {
          Cf[idx] = v;
        } else if (mode == 1) {
          float sc = rowScale ? rowScale[(size_t)row * rss] : 1.f;
          float prev = accumulate ? Cf[idx] : 0.f;
          Cf[idx] = prev + sc * v;
        } else if (mode == 2) {
          Ch[idx] = (f16)gelu_exact(v);
        } else {
          Ch[idx] = (f16)v;
        }
      }
    }
  }
}

// ---------------- small kernels ----------------
__global__ void cvt_f16_kernel(const float* __restrict__ x, f16* __restrict__ y, int n) {
  int i = blockIdx.x * 256 + threadIdx.x;
  if (i < n) y[i] = (f16)x[i];
}

// transpose-convert weight W[K][N] fp32 -> Wt[N][K] f16 (so B fragments load contiguously)
__global__ void convert_wT(const float* __restrict__ W, f16* __restrict__ Wt, int K, int N) {
  int i = blockIdx.x * 256 + threadIdx.x;
  if (i >= K * N) return;
  int k = i / N, n = i - k * N;
  Wt[(size_t)n * K + k] = (f16)W[i];
}

// attention over memory: r[b] = softmax(q[b]·mk[b,s,:]/sqrt(D)) @ mv[b]
__global__ void attn_kernel(const float* __restrict__ q, const float* __restrict__ mk,
                            const float* __restrict__ mv, float* __restrict__ r) {
  __shared__ float qs[DD];
  __shared__ float att[SS];
  __shared__ float invs;
  int b = blockIdx.x;
  int tid = threadIdx.x;
  for (int d = tid; d < DD; d += 256) qs[d] = q[(size_t)b * DD + d];
  __syncthreads();
  int wave = tid >> 5, lane = tid & 31;
  const float scale = 0.044194173824159216f;  // 1/sqrt(512)
  for (int i = 0; i < 16; ++i) {
    int s = wave * 16 + i;
    const float* row = mk + ((size_t)b * SS + s) * DD;
    float acc = 0.f;
    for (int d = lane; d < DD; d += 32) acc += qs[d] * row[d];
    acc = wsum32(acc);
    if (lane == 0) att[s] = acc * scale;
  }
  __syncthreads();
  if (wave == 0) {
    float m = -1e30f;
    for (int i = lane; i < SS; i += 32) m = fmaxf(m, att[i]);
    m = wmax32(m);
    float sum = 0.f;
    for (int i = lane; i < SS; i += 32) { float e = __expf(att[i] - m); att[i] = e; sum += e; }
    sum = wsum32(sum);
    if (lane == 0) invs = 1.f / sum;
  }
  __syncthreads();
  float iv = invs;
  for (int d = tid; d < DD; d += 256) {
    float acc = 0.f;
    const float* base = mv + (size_t)b * SS * DD + d;
    for (int s = 0; s < SS; ++s) acc += att[s] * base[(size_t)s * DD];
    r[(size_t)b * DD + d] = acc * iv;
  }
}

// per-row router logits (exp-shifted) + mix softmax; wave per row
__global__ void rowops_kernel(const float* __restrict__ s,
                              const float* __restrict__ router_w, const float* __restrict__ router_b,
                              const float* __restrict__ mix_w, const float* __restrict__ mix_b,
                              float* __restrict__ P, float* __restrict__ mixw) {
  int wave = threadIdx.x >> 5, lane = threadIdx.x & 31;
  int b = blockIdx.x * 8 + wave;
  if (b >= BB) return;
  const float* srow = s + (size_t)b * DD;
  float a0 = 0, a1 = 0, a2 = 0, a3 = 0, m0 = 0, m1 = 0, m2 = 0;
  for (int d = lane; d < DD; d += 32) {
    float sv = srow[d];
    a0 += sv * router_w[d * EE + 0];
    a1 += sv * router_w[d * EE + 1];
    a2 += sv * router_w[d * EE + 2];
    a3 += sv * router_w[d * EE + 3];
    m0 += sv * mix_w[d * 3 + 0];
    m1 += sv * mix_w[d * 3 + 1];
    m2 += sv * mix_w[d * 3 + 2];
  }
  a0 = wsum32(a0); a1 = wsum32(a1); a2 = wsum32(a2); a3 = wsum32(a3);
  m0 = wsum32(m0); m1 = wsum32(m1); m2 = wsum32(m2);
  if (lane == 0) {
    float l0 = a0 + router_b[0], l1 = a1 + router_b[1];
    float l2 = a2 + router_b[2], l3 = a3 + router_b[3];
    float mx = fmaxf(fmaxf(l0, l1), fmaxf(l2, l3));
    P[b * 4 + 0] = __expf(l0 - mx) + EPS_;
    P[b * 4 + 1] = __expf(l1 - mx) + EPS_;
    P[b * 4 + 2] = __expf(l2 - mx) + EPS_;
    P[b * 4 + 3] = __expf(l3 - mx) + EPS_;
    float q0 = m0 + mix_b[0], q1 = m1 + mix_b[1], q2 = m2 + mix_b[2];
    float qm = fmaxf(q0, fmaxf(q1, q2));
    float e0 = __expf(q0 - qm), e1 = __expf(q1 - qm), e2 = __expf(q2 - qm);
    float inv = 1.f / (e0 + e1 + e2);
    mixw[b * 3 + 0] = e0 * inv;
    mixw[b * 3 + 1] = e1 * inv;
    mixw[b * 3 + 2] = e2 * inv;
  }
}

// Sinkhorn in LDS (single workgroup, needs global column sums over B) + top-1 coeff
__global__ void __launch_bounds__(1024) sinkhorn_kernel(const float* __restrict__ P,
                                                        float* __restrict__ psp) {
  __shared__ float Pl[BB * EE];   // 32 KB
  __shared__ float part[32][EE];
  __shared__ float cs[EE];
  int tid = threadIdx.x;
  for (int i = tid; i < BB * EE; i += 1024) Pl[i] = P[i];
  int r0 = tid * 2;
  const float colt = (float)BB / (float)EE;
  int wave = tid >> 5, lane = tid & 31;
  for (int it = 0; it < SINK_ITERS_; ++it) {
    __syncthreads();
    float c0 = 0, c1 = 0, c2 = 0, c3 = 0;
#pragma unroll
    for (int j = 0; j < 2; ++j) {
      int rr = r0 + j;
      float x0 = Pl[rr * 4 + 0], x1 = Pl[rr * 4 + 1];
      float x2 = Pl[rr * 4 + 2], x3 = Pl[rr * 4 + 3];
      float f = 1.f / (x0 + x1 + x2 + x3 + EPS_);
      x0 *= f; x1 *= f; x2 *= f; x3 *= f;
      Pl[rr * 4 + 0] = x0; Pl[rr * 4 + 1] = x1;
      Pl[rr * 4 + 2] = x2; Pl[rr * 4 + 3] = x3;
      c0 += x0; c1 += x1; c2 += x2; c3 += x3;
    }
    c0 = wsum32(c0); c1 = wsum32(c1); c2 = wsum32(c2); c3 = wsum32(c3);
    if (lane == 0) { part[wave][0] = c0; part[wave][1] = c1; part[wave][2] = c2; part[wave][3] = c3; }
    __syncthreads();
    if (tid < EE) {
      float t = 0.f;
      for (int w = 0; w < 32; ++w) t += part[w][tid];
      cs[tid] = colt / (t + EPS_);
    }
    __syncthreads();
#pragma unroll
    for (int j = 0; j < 2; ++j) {
      int rr = r0 + j;
      Pl[rr * 4 + 0] *= cs[0]; Pl[rr * 4 + 1] *= cs[1];
      Pl[rr * 4 + 2] *= cs[2]; Pl[rr * 4 + 3] *= cs[3];
    }
  }
  __syncthreads();
#pragma unroll
  for (int j = 0; j < 2; ++j) {
    int rr = r0 + j;
    float x0 = Pl[rr * 4 + 0], x1 = Pl[rr * 4 + 1];
    float x2 = Pl[rr * 4 + 2], x3 = Pl[rr * 4 + 3];
    float f = 1.f / (x0 + x1 + x2 + x3 + EPS_);
    x0 *= f; x1 *= f; x2 *= f; x3 *= f;
    int am = 0; float bv = x0;
    if (x1 > bv) { bv = x1; am = 1; }
    if (x2 > bv) { bv = x2; am = 2; }
    if (x3 > bv) { bv = x3; am = 3; }
    float coef = bv / (bv + 1e-8f);
    psp[rr * 4 + 0] = (am == 0) ? coef : 0.f;
    psp[rr * 4 + 1] = (am == 1) ? coef : 0.f;
    psp[rr * 4 + 2] = (am == 2) ? coef : 0.f;
    psp[rr * 4 + 3] = (am == 3) ? coef : 0.f;
  }
}

__global__ void cat_kernel(const float* __restrict__ s, const float* __restrict__ r,
                           f16* __restrict__ cat16) {
  int i = blockIdx.x * 256 + threadIdx.x;
  if (i >= BB * 2 * DD) return;
  int b = i / (2 * DD), d = i - b * 2 * DD;
  float v = (d < DD) ? s[(size_t)b * DD + d] : r[(size_t)b * DD + d - DD];
  cat16[i] = (f16)v;
}

__global__ void combine_kernel(const float* __restrict__ s, const float* __restrict__ hb,
                               const float* __restrict__ hm, const float* __restrict__ hmem,
                               const float* __restrict__ mixw,
                               float* __restrict__ snew, f16* __restrict__ snew16) {
  int i = blockIdx.x * 256 + threadIdx.x;
  if (i >= BB * DD) return;
  int b = i / DD;
  float m0 = mixw[b * 3 + 0], m1 = mixw[b * 3 + 1], m2 = mixw[b * 3 + 2];
  float v = s[i] + ETA_ * (m0 * hb[i] + m1 * hm[i] + m2 * hmem[i]);
  snew[i] = v;
  snew16[i] = (f16)v;
}

// RMS-normalize wvec (f16 out for wk/wv GEMMs) + scalar gate per row; wave per row
__global__ void rmsgate_kernel(const float* __restrict__ wvec, const float* __restrict__ snew,
                               const float* __restrict__ gate_w, const float* __restrict__ gate_b,
                               f16* __restrict__ wvec16, float* __restrict__ g) {
  int wave = threadIdx.x >> 5, lane = threadIdx.x & 31;
  int b = blockIdx.x * 8 + wave;
  if (b >= BB) return;
  const float* w = wvec + (size_t)b * DD;
  const float* sr = snew + (size_t)b * DD;
  float ss = 0.f, gd = 0.f;
  for (int d = lane; d < DD; d += 32) {
    float v = w[d];
    ss += v * v;
    gd += sr[d] * gate_w[d];
  }
  ss = wsum32(ss);
  gd = wsum32(gd);
  float sc = rsqrtf(ss * (1.f / DD) + EPS_);
  for (int d = lane; d < DD; d += 32) wvec16[(size_t)b * DD + d] = (f16)(w[d] * sc);
  if (lane == 0) g[b] = GATE_MAX_ / (1.f + __expf(-(gd + gate_b[0])));
}

// softmax over S=128 write-weights; wave per row
__global__ void softmax128_kernel(float* __restrict__ ww) {
  int wave = threadIdx.x >> 5, lane = threadIdx.x & 31;
  int b = blockIdx.x * 8 + wave;
  if (b >= BB) return;
  float* row = ww + (size_t)b * SS;
  float v0 = row[lane], v1 = row[lane + 32], v2 = row[lane + 64], v3 = row[lane + 96];
  float m = wmax32(fmaxf(fmaxf(v0, v1), fmaxf(v2, v3)));
  v0 = __expf(v0 - m); v1 = __expf(v1 - m); v2 = __expf(v2 - m); v3 = __expf(v3 - m);
  float inv = 1.f / wsum32(v0 + v1 + v2 + v3);
  row[lane] = v0 * inv; row[lane + 32] = v1 * inv;
  row[lane + 64] = v2 * inv; row[lane + 96] = v3 * inv;
}

// EMA memory update (bandwidth critical): mk += g*ww*(wk - mk), mv += g*ww*(wv - mv)
__global__ void update_kernel(float* __restrict__ mk, float* __restrict__ mv,
                              const float* __restrict__ wk, const float* __restrict__ wv,
                              const float* __restrict__ ww, const float* __restrict__ g) {
  size_t i = (size_t)blockIdx.x * 256 + threadIdx.x;
  const size_t total = (size_t)BB * SS * (DD / 4);
  if (i >= total) return;
  const int perB = SS * (DD / 4);
  int b = (int)(i / perB);
  int rem = (int)(i - (size_t)b * perB);
  int s = rem / (DD / 4);
  int d4 = rem - s * (DD / 4);
  float gg = g[b] * ww[(size_t)b * SS + s];
  const float4* wk4 = (const float4*)wk;
  const float4* wv4 = (const float4*)wv;
  float4 k4 = wk4[(size_t)b * (DD / 4) + d4];
  float4 v4 = wv4[(size_t)b * (DD / 4) + d4];
  float4* mk4 = (float4*)mk;
  float4* mv4 = (float4*)mv;
  float4 a = mk4[i];
  float4 c = mv4[i];
  a.x += gg * (k4.x - a.x); a.y += gg * (k4.y - a.y);
  a.z += gg * (k4.z - a.z); a.w += gg * (k4.w - a.w);
  c.x += gg * (v4.x - c.x); c.y += gg * (v4.y - c.y);
  c.z += gg * (v4.z - c.z); c.w += gg * (v4.w - c.w);
  mk4[i] = a;
  mv4[i] = c;
}

// ---------------- host orchestration ----------------
extern "C" void kernel_launch(void* const* d_in, const int* in_sizes, int n_in,
                              void* d_out, int out_size, void* d_ws, size_t ws_size,
                              hipStream_t stream) {
  (void)in_sizes; (void)n_in; (void)out_size; (void)ws_size;
  const float* s0      = (const float*)d_in[0];
  const float* mem_k0  = (const float*)d_in[1];
  const float* mem_v0  = (const float*)d_in[2];
  const float* q_w     = (const float*)d_in[3];
  const float* wl_w    = (const float*)d_in[4];
  const float* wl_b    = (const float*)d_in[5];
  const float* wk_w    = (const float*)d_in[6];
  const float* wv_w    = (const float*)d_in[7];
  const float* wvec_w  = (const float*)d_in[8];
  const float* base_w1 = (const float*)d_in[9];
  const float* base_w2 = (const float*)d_in[10];
  const float* mem_w1  = (const float*)d_in[11];
  const float* mem_w2  = (const float*)d_in[12];
  const float* router_w = (const float*)d_in[13];
  const float* router_b = (const float*)d_in[14];
  const float* exp_w1  = (const float*)d_in[15];
  const float* exp_w2  = (const float*)d_in[16];
  const float* mix_w   = (const float*)d_in[17];
  const float* mix_b   = (const float*)d_in[18];
  const float* gate_w  = (const float*)d_in[19];
  const float* gate_b  = (const float*)d_in[20];

  char* base = (char*)d_ws;
  size_t off = 0;
  auto alloc = [&](size_t bytes) -> char* {
    char* p = base + off;
    off = (off + bytes + 255) & ~(size_t)255;
    return p;
  };
  float* mkw   = (float*)alloc((size_t)BB * SS * DD * 4);
  float* mvw   = (float*)alloc((size_t)BB * SS * DD * 4);
  float* sA    = (float*)alloc((size_t)BB * DD * 4);
  float* sB    = (float*)alloc((size_t)BB * DD * 4);
  f16*   s16   = (f16*)alloc((size_t)BB * DD * 2);
  float* qv    = (float*)alloc((size_t)BB * DD * 4);
  float* rv    = (float*)alloc((size_t)BB * DD * 4);
  f16*   cat16 = (f16*)alloc((size_t)BB * 2 * DD * 2);
  f16*   hid16 = (f16*)alloc((size_t)BB * HH * 2);
  float* hb    = (float*)alloc((size_t)BB * DD * 4);
  float* hm    = (float*)alloc((size_t)BB * DD * 4);
  float* hmem  = (float*)alloc((size_t)BB * DD * 4);
  float* P     = (float*)alloc((size_t)BB * EE * 4);
  float* psp   = (float*)alloc((size_t)BB * EE * 4);
  float* mixw  = (float*)alloc((size_t)BB * 3 * 4);
  float* g     = (float*)alloc((size_t)BB * 4);
  float* wvec  = (float*)alloc((size_t)BB * DD * 4);
  f16*   wvec16 = (f16*)alloc((size_t)BB * DD * 2);
  float* ww    = (float*)alloc((size_t)BB * SS * 4);
  float* wk    = (float*)alloc((size_t)BB * DD * 4);
  float* wv    = (float*)alloc((size_t)BB * DD * 4);
  f16* qwt  = (f16*)alloc((size_t)DD * DD * 2);
  f16* wlt  = (f16*)alloc((size_t)DD * SS * 2);
  f16* wkt  = (f16*)alloc((size_t)DD * DD * 2);
  f16* wvt  = (f16*)alloc((size_t)DD * DD * 2);
  f16* wvwt = (f16*)alloc((size_t)DD * DD * 2);
  f16* bw1t = (f16*)alloc((size_t)DD * HH * 2);
  f16* bw2t = (f16*)alloc((size_t)HH * DD * 2);
  f16* mw1t = (f16*)alloc((size_t)2 * DD * HH * 2);
  f16* mw2t = (f16*)alloc((size_t)HH * DD * 2);
  f16* ew1t = (f16*)alloc((size_t)EE * DD * HH * 2);
  f16* ew2t = (f16*)alloc((size_t)EE * HH * DD * 2);

  auto cdiv = [](size_t a, size_t b) { return (int)((a + b - 1) / b); };
  auto convT = [&](const float* W, f16* Wt, int K, int N) {
    convert_wT<<<cdiv((size_t)K * N, 256), 256, 0, stream>>>(W, Wt, K, N);
  };
  auto gemm = [&](const f16* A, const f16* Bt, float* Cf, f16* Ch, const float* bias,
                  const float* rs, int rss, int M, int N, int K, int mode, int accum) {
    int tiles = (M / 32) * (N / 32);
    gemm_wmma<<<(tiles + 7) / 8, 256, 0, stream>>>(A, Bt, Cf, Ch, bias, rs, rss,
                                                   M, N, K, mode, accum);
  };

  // private copies of the recurrent memory (inputs must not be mutated)
  hipMemcpyAsync(mkw, mem_k0, (size_t)BB * SS * DD * 4, hipMemcpyDeviceToDevice, stream);
  hipMemcpyAsync(mvw, mem_v0, (size_t)BB * SS * DD * 4, hipMemcpyDeviceToDevice, stream);

  // one-time transposed f16 weight conversion
  convT(q_w, qwt, DD, DD);
  convT(wl_w, wlt, DD, SS);
  convT(wk_w, wkt, DD, DD);
  convT(wv_w, wvt, DD, DD);
  convT(wvec_w, wvwt, DD, DD);
  convT(base_w1, bw1t, DD, HH);
  convT(base_w2, bw2t, HH, DD);
  convT(mem_w1, mw1t, 2 * DD, HH);
  convT(mem_w2, mw2t, HH, DD);
  for (int e = 0; e < EE; ++e) {
    convT(exp_w1 + (size_t)e * DD * HH, ew1t + (size_t)e * DD * HH, DD, HH);
    convT(exp_w2 + (size_t)e * HH * DD, ew2t + (size_t)e * HH * DD, HH, DD);
  }

  for (int k = 0; k < KSTEPS; ++k) {
    const float* sCur = (k == 0) ? s0 : ((k & 1) ? sA : sB);
    float* sNext = (k & 1) ? sB : sA;

    // s -> f16 for GEMM A-operand
    cvt_f16_kernel<<<cdiv((size_t)BB * DD, 256), 256, 0, stream>>>(sCur, s16, BB * DD);
    // q = s @ q_w
    gemm(s16, qwt, qv, nullptr, nullptr, nullptr, 0, BB, DD, DD, 0, 0);
    // attention read over memory
    attn_kernel<<<BB, 256, 0, stream>>>(qv, mkw, mvw, rv);
    // cat = [s, r] in f16
    cat_kernel<<<cdiv((size_t)BB * 2 * DD, 256), 256, 0, stream>>>(sCur, rv, cat16);
    // base MLP
    gemm(s16, bw1t, nullptr, hid16, nullptr, nullptr, 0, BB, HH, DD, 2, 0);
    gemm(hid16, bw2t, hb, nullptr, nullptr, nullptr, 0, BB, DD, HH, 0, 0);
    // router + mix logits, then Sinkhorn + top-1 coefficients
    rowops_kernel<<<BB / 8, 256, 0, stream>>>(sCur, router_w, router_b, mix_w, mix_b, P, mixw);
    sinkhorn_kernel<<<1, 1024, 0, stream>>>(P, psp);
    // dense MoE: h_moe = sum_e psp[:,e] * (gelu(s@w1_e)@w2_e)  (exact: off-experts weight 0)
    for (int e = 0; e < EE; ++e) {
      gemm(s16, ew1t + (size_t)e * DD * HH, nullptr, hid16, nullptr, nullptr, 0,
           BB, HH, DD, 2, 0);
      gemm(hid16, ew2t + (size_t)e * HH * DD, hm, nullptr, nullptr, psp + e, EE,
           BB, DD, HH, 1, (e > 0) ? 1 : 0);
    }
    // memory MLP on [s, r]
    gemm(cat16, mw1t, nullptr, hid16, nullptr, nullptr, 0, BB, HH, 2 * DD, 2, 0);
    gemm(hid16, mw2t, hmem, nullptr, nullptr, nullptr, 0, BB, DD, HH, 0, 0);
    // s_new = s + eta * mix-weighted candidates (also emits f16 copy)
    combine_kernel<<<cdiv((size_t)BB * DD, 256), 256, 0, stream>>>(sCur, hb, hm, hmem,
                                                                   mixw, sNext, s16);
    // wvec = rmsnorm(s_new @ wvec_w); gate
    gemm(s16, wvwt, wvec, nullptr, nullptr, nullptr, 0, BB, DD, DD, 0, 0);
    rmsgate_kernel<<<BB / 8, 256, 0, stream>>>(wvec, sNext, gate_w, gate_b, wvec16, g);
    // write weights ww = softmax(s_new @ wl_w + wl_b)
    gemm(s16, wlt, ww, nullptr, wl_b, nullptr, 0, BB, SS, DD, 0, 0);
    softmax128_kernel<<<BB / 8, 256, 0, stream>>>(ww);
    // wk / wv projections of normalized wvec
    gemm(wvec16, wkt, wk, nullptr, nullptr, nullptr, 0, BB, DD, DD, 0, 0);
    gemm(wvec16, wvt, wv, nullptr, nullptr, nullptr, 0, BB, DD, DD, 0, 0);
    // EMA memory update (bandwidth dominant)
    update_kernel<<<cdiv((size_t)BB * SS * (DD / 4), 256), 256, 0, stream>>>(mkw, mvw, wk, wv,
                                                                             ww, g);
  }

  // K=4 (even) -> final state in sB
  hipMemcpyAsync(d_out, sB, (size_t)BB * DD * 4, hipMemcpyDeviceToDevice, stream);
}
